// SNNLayer_16612933501495
// MI455X (gfx1250) — compile-verified, matrix operand order checked by reading, not run on hardware
//
#include <hip/hip_runtime.h>
#include <hip/hip_bf16.h>

// ---------------------------------------------------------------------------
// SNN layer for MI455X (gfx1250, wave32, WMMA 16x16x32 f16, TDM async loads)
//
//   phase 1: syn = depthwise-conv(x)                 (mem-bound, f16 out)
//   phase 2: feed = syn @ kernel + bias              (WMMA GEMM, 64x64/wave,
//                                                     double-buffered frags)
//   phase 3: per-16-batch-row scan (no cross-WG sync needed):
//              out = state@rec + feed[:,t,:]          rec in VGPRs,
//              n = sigmoid(3*out)                     state in LDS,
//              state = out*sigmoid(1-2n)              feed staged by TDM
// ---------------------------------------------------------------------------

typedef __attribute__((ext_vector_type(16))) _Float16 v16h;
typedef __attribute__((ext_vector_type(8)))  float    v8f;
typedef __attribute__((ext_vector_type(4)))  unsigned int u32x4;
typedef __attribute__((ext_vector_type(8)))  int      i32x8;
typedef __attribute__((ext_vector_type(4)))  int      i32x4;
typedef _Float16 half_t;

#define B_   256
#define T_   1024
#define W_   512
#define U_   512
#define P_   16
#define TP_  (T_ - P_ + 1)        // 1009
#define MTOT (B_ * TP_)           // 258304 (= 64 * 4036)
#define NKT_ (W_ / 32)            // 16 k-tiles of 32 along K=512

union AFrag { uint4 u[2]; v16h v; };

// ---------------------------------------------------------------------------
// Pack a (K x N) f32 row-major matrix into WMMA-B fragment order (f16).
// Fragment (ntile,ktile): 32 lanes x 16 halves contiguous.
//   lane<16 : n = lane,    k = k0 + {0..7, 16..23}
//   lane>=16: n = lane-16, k = k0 + {8..15, 24..31}
// ---------------------------------------------------------------------------
__global__ void pack_b_kernel(const float* __restrict__ src,
                              half_t* __restrict__ dst, int K, int N) {
  int gid  = blockIdx.x * blockDim.x + threadIdx.x;
  int lane = gid & 31;
  int tile = gid >> 5;
  int nkt  = K >> 5;
  int ntile = tile / nkt;
  int ktile = tile - ntile * nkt;
  if (ntile >= (N >> 4)) return;
  int n  = (ntile << 4) + (lane & 15);
  int k0 = (ktile << 5) + ((lane & 16) ? 8 : 0);
  half_t* out = dst + ((size_t)tile * 32 + lane) * 16;
#pragma unroll
  for (int h = 0; h < 16; ++h) {
    int k = k0 + (h < 8 ? h : h + 8);
    out[h] = (half_t)src[(size_t)k * N + n];
  }
}

// ---------------------------------------------------------------------------
// Depthwise temporal conv: syn[b,t,w] = pb + sum_k pw[k]*x[b,t+k,w]  (f16 out)
// One thread per (b, w/4); register ring buffer, statically unrolled indices.
// ---------------------------------------------------------------------------
__global__ void __launch_bounds__(256)
psc_conv_kernel(const float* __restrict__ x, const float* __restrict__ pw,
                const float* __restrict__ pb, half_t* __restrict__ syn) {
  int gid = blockIdx.x * blockDim.x + threadIdx.x;   // B_*(W_/4) = 32768
  int w4  = gid & (W_ / 4 - 1);
  int b   = gid >> 7;
  if (b >= B_) return;

  float wk[P_];
#pragma unroll
  for (int k = 0; k < P_; ++k) wk[k] = pw[k];
  const float bias = pb[0];

  const float4* xp = (const float4*)(x + (size_t)b * T_ * W_) + w4;
  half_t*       sp = syn + (size_t)b * TP_ * W_ + (size_t)w4 * 4;

  float4 ring[P_];

  auto emit = [&](int t, int j) {   // t&15 == j (tb multiple of 16)
    float4 acc = {bias, bias, bias, bias};
#pragma unroll
    for (int k = 0; k < P_; ++k) {
      float4 v = ring[(j + 1 + k) & 15];
      float  c = wk[k];
      acc.x += c * v.x; acc.y += c * v.y; acc.z += c * v.z; acc.w += c * v.w;
    }
    half_t* o = sp + (size_t)(t - (P_ - 1)) * W_;
    o[0] = (half_t)acc.x; o[1] = (half_t)acc.y;
    o[2] = (half_t)acc.z; o[3] = (half_t)acc.w;
  };

#pragma unroll
  for (int j = 0; j < 16; ++j) ring[j] = xp[(size_t)j * (W_ / 4)];
  emit(15, 15);
  for (int tb = 16; tb < T_; tb += 16) {
#pragma unroll
    for (int j = 0; j < 16; ++j) {
      int t = tb + j;
      ring[j] = xp[(size_t)t * (W_ / 4)];
      emit(t, j);
    }
  }
}

// ---------------------------------------------------------------------------
// feed = syn @ kernel + bias   (M=MTOT, N=512, K=512), f32 out in C-layout.
// 8 waves/WG. Each wave: 64(M) x 64(N) macro-tile (4x4 WMMA tiles),
// K looped in 32-chunks with register double-buffering of A/B fragments.
// ---------------------------------------------------------------------------
__global__ void __launch_bounds__(256)
gemm_feed_kernel(const half_t* __restrict__ syn,
                 const half_t* __restrict__ packK,
                 const float* __restrict__ bias,
                 float* __restrict__ feed) {
  int m0    = blockIdx.x * 64;           // 4036 blocks
  int wave  = threadIdx.x >> 5;          // 0..7 -> N strip of 64
  int lane  = threadIdx.x & 31;
  int rlane = lane & 15;
  int koff  = (lane & 16) ? 8 : 0;
  int hi    = lane >> 4;
  int nb    = wave * 64;

  const half_t* abase[4];
#pragma unroll
  for (int mi = 0; mi < 4; ++mi)
    abase[mi] = syn + (size_t)(m0 + mi * 16 + rlane) * W_ + koff;

  auto loadA = [&](AFrag (&a)[4], int kt) {
#pragma unroll
    for (int mi = 0; mi < 4; ++mi) {
      const half_t* ap = abase[mi] + kt * 32;
      a[mi].u[0] = *(const uint4*)ap;
      a[mi].u[1] = *(const uint4*)(ap + 16);
    }
  };
  auto loadB = [&](v16h (&bf)[4], int kt) {
#pragma unroll
    for (int j = 0; j < 4; ++j) {
      int ntile = (nb >> 4) + j;
      bf[j] = *(const v16h*)(packK +
          (((size_t)ntile * NKT_ + kt) * 32 + lane) * 16);
    }
  };

  v8f acc[4][4] = {};
  AFrag a_cur[4], a_nxt[4];
  v16h  b_cur[4], b_nxt[4];
  loadA(a_cur, 0);
  loadB(b_cur, 0);
#pragma unroll
  for (int kt = 0; kt < NKT_; ++kt) {
    if (kt + 1 < NKT_) {        // prefetch next chunk while 16 WMMAs run
      loadA(a_nxt, kt + 1);
      loadB(b_nxt, kt + 1);
    }
#pragma unroll
    for (int mi = 0; mi < 4; ++mi)
#pragma unroll
      for (int j = 0; j < 4; ++j)
        acc[mi][j] = __builtin_amdgcn_wmma_f32_16x16x32_f16(
            false, a_cur[mi].v, false, b_cur[j], (short)0, acc[mi][j],
            false, false);
#pragma unroll
    for (int mi = 0; mi < 4; ++mi) a_cur[mi] = a_nxt[mi];
#pragma unroll
    for (int j = 0; j < 4; ++j) b_cur[j] = b_nxt[j];
  }

#pragma unroll
  for (int j = 0; j < 4; ++j) {
    int col = nb + j * 16 + rlane;
    float bv = bias[col];
#pragma unroll
    for (int mi = 0; mi < 4; ++mi)
#pragma unroll
      for (int r = 0; r < 8; ++r) {
        int m = m0 + mi * 16 + r + 8 * hi;
        feed[(size_t)m * U_ + col] = acc[mi][j][r] + bv;
      }
  }
}

// ---------------------------------------------------------------------------
// TDM: async-load one step's feed tile (16 batch rows x 512 f32, row stride
// TP_*U_ elements) into LDS. D# built per CDNA5 ISA.
// This toolchain exposes the 6-arg builtin:
//   (u32x4 g0, i32x8 g1, i32x4 g2, i32x4 g3, i32x8 ext, i32 cpol)
// ---------------------------------------------------------------------------
static __device__ __forceinline__ void tdm_feed_load(const float* gsrc,
                                                     unsigned lds_off) {
  unsigned long long ga = (unsigned long long)(size_t)gsrc;
  u32x4 g0;
  g0[0] = 1u;                                    // count=1 (valid), user mode
  g0[1] = lds_off;                               // lds_addr (bytes)
  g0[2] = (unsigned)(ga & 0xffffffffu);          // global_addr[31:0]
  g0[3] = (unsigned)((ga >> 32) & 0x01ffffffu)   // global_addr[56:32]
          | (2u << 30);                          // type = 2 ("image")
  i32x8 g1;
  g1[0] = (int)(2u << 16);                       // data_size = 4 bytes
  g1[1] = (int)((unsigned)W_ << 16);             // tensor_dim0 = 512 (lo16)
  g1[2] = (int)(16u << 16);                      // tdim0 hi | tensor_dim1 = 16
  g1[3] = (int)((unsigned)W_ << 16);             // tdim1 hi | tile_dim0 = 512
  g1[4] = 16;                                    // tile_dim1 = 16, tile_dim2=0
  g1[5] = (int)((unsigned)(TP_ * U_));           // tensor_dim0_stride lo32
  g1[6] = 0;                                     // stride hi16 | tdim1_stride
  g1[7] = 0;
  i32x4 z4 = {0, 0, 0, 0};
  i32x8 z8 = {0, 0, 0, 0, 0, 0, 0, 0};
  __builtin_amdgcn_tensor_load_to_lds(g0, g1, z4, z4, z8, 0);
}

// ---------------------------------------------------------------------------
// Sequential scan. One WG per 16 batch rows (16 WGs total, independent).
// 16 waves/WG; wave owns 2 N-tiles; rec B-fragments resident in VGPRs
// (2*16 v16h = 256 VGPRs); 16x512 f16 state tile resident in LDS; feed
// tiles double-buffered in LDS via TDM, one step ahead of compute.
// ---------------------------------------------------------------------------
#define SROW 520   // 512 + 8 halves pad

__global__ void __launch_bounds__(512)
snn_scan_kernel(const float* __restrict__ feed,
                const half_t* __restrict__ packR,
                float* __restrict__ out) {
  __shared__ __align__(16) half_t st[16][SROW];
  __shared__ __align__(16) float  fbuf[2][16][W_];   // TDM staging, 64 KB

  int bb   = blockIdx.x << 4;           // batch row base
  int wave = threadIdx.x >> 5;          // 0..15
  int lane = threadIdx.x & 31;
  int rowA = lane & 15;
  int koff = (lane & 16) ? 8 : 0;
  int hi   = lane >> 4;

  // rec B-fragments -> registers (zero B traffic during the scan)
  v16h Br[2][NKT_];
#pragma unroll
  for (int j = 0; j < 2; ++j) {
    int ntile = wave * 2 + j;
#pragma unroll
    for (int kt = 0; kt < NKT_; ++kt)
      Br[j][kt] = *(const v16h*)(packR +
          (((size_t)ntile * NKT_ + kt) * 32 + lane) * 16);
  }

  const float* fstep = feed + (size_t)bb * TP_ * U_;   // +t*U_ per step
  unsigned lds_f[2];
  lds_f[0] = (unsigned)(size_t)&fbuf[0][0][0];
  lds_f[1] = (unsigned)(size_t)&fbuf[1][0][0];

  // kick off step-0 feed tile
  if (threadIdx.x < 32) tdm_feed_load(fstep, lds_f[0]);

  // state0 = 0
  for (int i = threadIdx.x; i < 16 * SROW; i += 512)
    ((half_t*)st)[i] = (half_t)0.f;

  if (threadIdx.x < 32) __builtin_amdgcn_s_wait_tensorcnt(0);
  __syncthreads();

  for (int t = 0; t < TP_; ++t) {
    // async-stage next step's feed tile while this step computes
    if (threadIdx.x < 32 && t + 1 < TP_)
      tdm_feed_load(fstep + (size_t)(t + 1) * U_, lds_f[(t + 1) & 1]);

    const float* fb = &fbuf[t & 1][0][0];
    v8f acc[2];
#pragma unroll
    for (int j = 0; j < 2; ++j) {
      int col = wave * 32 + j * 16 + rowA;
#pragma unroll
      for (int r = 0; r < 8; ++r)
        acc[j][r] = fb[(r + 8 * hi) * W_ + col];   // ds_load from staged tile
    }

    // out = state @ rec (+feed in C); A-fragments one-ahead from LDS
    AFrag a_c, a_n;
    {
      const half_t* ap = &st[rowA][koff];
      a_c.u[0] = *(const uint4*)ap;
      a_c.u[1] = *(const uint4*)(ap + 16);
    }
#pragma unroll
    for (int kt = 0; kt < NKT_; ++kt) {
      if (kt + 1 < NKT_) {
        const half_t* ap = &st[rowA][(kt + 1) * 32 + koff];
        a_n.u[0] = *(const uint4*)ap;
        a_n.u[1] = *(const uint4*)(ap + 16);
      }
      acc[0] = __builtin_amdgcn_wmma_f32_16x16x32_f16(
          false, a_c.v, false, Br[0][kt], (short)0, acc[0], false, false);
      acc[1] = __builtin_amdgcn_wmma_f32_16x16x32_f16(
          false, a_c.v, false, Br[1][kt], (short)0, acc[1], false, false);
      a_c = a_n;
    }
    __syncthreads();   // all waves done reading state

    // n = sigmoid(3*out); state = out * sigmoid(1-2n)
#pragma unroll
    for (int j = 0; j < 2; ++j) {
      int col = wave * 32 + j * 16 + rowA;
#pragma unroll
      for (int r = 0; r < 8; ++r) {
        float o = acc[j][r];
        float n = 1.f / (1.f + __expf(-3.f * o));
        int b = bb + r + 8 * hi;
        out[((size_t)b * TP_ + t) * U_ + col] = n;
        float g = 1.f / (1.f + __expf(2.f * n - 1.f)); // sigmoid(1-2n)
        st[r + 8 * hi][col] = (half_t)(o * g);
      }
    }
    if (threadIdx.x < 32)
      __builtin_amdgcn_s_wait_tensorcnt(0);   // next feed tile landed
    __syncthreads();   // new state + staged feed visible to all waves
  }
}

// ---------------------------------------------------------------------------
extern "C" void kernel_launch(void* const* d_in, const int* in_sizes, int n_in,
                              void* d_out, int out_size, void* d_ws,
                              size_t ws_size, hipStream_t stream) {
  const float* x     = (const float*)d_in[0];   // (B,T,W)
  const float* psc_w = (const float*)d_in[1];   // (P,)
  const float* psc_b = (const float*)d_in[2];   // (1,)
  const float* kernW = (const float*)d_in[3];   // (W,U)
  const float* recW  = (const float*)d_in[4];   // (U,U)
  const float* bias  = (const float*)d_in[5];   // (U,)
  float* out = (float*)d_out;                   // (B,Tp,U)

  char* ws = (char*)d_ws;
  size_t off = 0;
  auto take = [&](size_t bytes) {
    void* p = ws + off;
    off = (off + bytes + 255) & ~(size_t)255;
    return p;
  };
  half_t* syn   = (half_t*)take((size_t)MTOT * W_ * sizeof(half_t)); // 264 MB
  half_t* packK = (half_t*)take((size_t)W_ * U_ * sizeof(half_t));   // 0.5 MB
  half_t* packR = (half_t*)take((size_t)U_ * U_ * sizeof(half_t));   // 0.5 MB
  float*  feed  = (float*)take((size_t)MTOT * U_ * sizeof(float));   // 529 MB
  (void)ws_size; (void)n_in; (void)in_sizes; (void)out_size;

  {
    int tiles = (W_ / 32) * (U_ / 16);          // 512
    int thr = tiles * 32;                       // 16384
    pack_b_kernel<<<thr / 256, 256, 0, stream>>>(kernW, packK, W_, U_);
    pack_b_kernel<<<thr / 256, 256, 0, stream>>>(recW,  packR, U_, U_);
  }
  psc_conv_kernel<<<(B_ * (W_ / 4)) / 256, 256, 0, stream>>>(x, psc_w, psc_b,
                                                             syn);
  gemm_feed_kernel<<<MTOT / 64, 256, 0, stream>>>(syn, packK, bias, feed);
  snn_scan_kernel<<<B_ / 16, 512, 0, stream>>>(feed, packR, out);
}
